// DeepSeekMLHAAttention_24051816858177
// MI455X (gfx1250) — matrix operand order repaired and checked by reading, not compiled, not run on hardware
//
#include <hip/hip_runtime.h>

// Problem constants (DeepSeek-style latent-KV GQA attention)
#define NH   16
#define NKV  4
#define HD   128
#define LAT  256
#define BB   2
#define TT   2048
#define HID  2048
#define MR   (BB*TT)      // 4096 token rows
#define KVD  (NKV*HD)     // 512

typedef __attribute__((ext_vector_type(8)))  float  v8f;
typedef __attribute__((ext_vector_type(8)))  __bf16 v8bf;
typedef __attribute__((ext_vector_type(16))) __bf16 v16bf;
typedef __attribute__((ext_vector_type(4)))  int    v4i;

// CDNA5 async global->LDS path (ASYNCcnt-tracked), with sync fallback.
#if __has_builtin(__builtin_amdgcn_global_load_async_to_lds_b128) && \
    __has_builtin(__builtin_amdgcn_s_wait_asynccnt)
#define ASYNC_LDS 1
#else
#define ASYNC_LDS 0
#endif

typedef __attribute__((address_space(1))) v4i* gv4i_p;   // global 16B chunk
typedef __attribute__((address_space(3))) v4i* sv4i_p;   // LDS    16B chunk

// Copy 16B from global to LDS for this lane.
__device__ __forceinline__ void copy16_g2s(const __bf16* g, __bf16* s) {
#if ASYNC_LDS
  __builtin_amdgcn_global_load_async_to_lds_b128(
      (gv4i_p)(void*)(__bf16*)g, (sv4i_p)(void*)s, 0, 0);
#else
  *(v8bf*)s = *(const v8bf*)g;
#endif
}
__device__ __forceinline__ void wait_async_le1() {
#if ASYNC_LDS
  __builtin_amdgcn_s_wait_asynccnt(1);
#endif
}
__device__ __forceinline__ void wait_async_le0() {
#if ASYNC_LDS
  __builtin_amdgcn_s_wait_asynccnt(0);
#endif
}

// ---------------------------------------------------------------------------
// A-operand fragment (16x32 bf16, M x K): lane = M row (lane&15),
// lanes 0-15 hold K {k0..k0+15}, lanes 16-31 hold {k0+16..k0+31}.
// Caller passes p already offset by (half*16); we grab [p,p+8) and [p+8,p+16).
__device__ __forceinline__ v16bf load_a_frag(const __bf16* p) {
  v8bf lo = *(const v8bf*)p;
  v8bf hi = *(const v8bf*)(p + 8);
  return __builtin_shufflevector(lo, hi, 0,1,2,3,4,5,6,7,8,9,10,11,12,13,14,15);
}
// B-operand fragment (32x16 bf16, K x N): lane = N col, per-lane K contiguous:
// lanes 0-15 hold K {k0..k0+15}, lanes 16-31 hold {k0+16..k0+31}. One 32B load.
__device__ __forceinline__ v16bf load_b_frag(const __bf16* p) {
  return *(const v16bf*)p;
}

// ---------------------------------------------------------------------------
__global__ void cvt_f32_to_bf16(const float* __restrict__ s,
                                __bf16* __restrict__ d, int n) {
  int i = blockIdx.x * blockDim.x + threadIdx.x;
  if (i < n) d[i] = (__bf16)s[i];
}

// ---------------------------------------------------------------------------
// C[M,N] (f32) = A[M,K] (bf16, row-major) * B[N,K]^T (bf16, row-major rows=N)
// Block: 256 threads = 8 waves arranged 2(M) x 4(N); wave tile 16x64.
// A block-tile (32 x 32K) is staged into LDS via async loads, double-buffered;
// B fragments stream straight from global (contiguous 32B per lane, L2-reused).
__global__ __launch_bounds__(256)
void gemm_bf16_nt(const __bf16* __restrict__ A, const __bf16* __restrict__ Bm,
                  float* __restrict__ C, int Mdim, int Ndim, int Kdim) {
  __shared__ __bf16 sA[2][32 * 32];      // 2 x 2KB K-slices of the A tile
  const int tid  = threadIdx.x;
  const int lane = tid & 31;
  const int wave = tid >> 5;
  const int half = lane >> 4;
  const int l16  = lane & 15;
  const int wm = wave >> 2, wn = wave & 3;
  const int mblock = blockIdx.x * 32;
  const int nbase  = blockIdx.y * 256 + wn * 64;

  // stagers: first 4 waves, each lane moves one 16B chunk (32 rows x 64B)
  const bool stager = tid < 128;
  const int  srow = tid >> 2;            // 0..31 for stagers
  const int  scol = (tid & 3) * 8;       // bf16 element offset of 16B chunk
  const __bf16* ag = A + (size_t)(mblock + srow) * Kdim + scol;

  if (stager) copy16_g2s(ag, &sA[0][srow * 32 + scol]);

  v8f acc[4] = {};
  const int nk = Kdim >> 5;
  for (int kt = 0; kt < nk; ++kt) {
    const int buf = kt & 1;
    if (kt + 1 < nk) {                   // prefetch next K-slice
      if (stager) copy16_g2s(ag + (size_t)(kt + 1) * 32,
                             &sA[buf ^ 1][srow * 32 + scol]);
      wait_async_le1();                  // previous slice landed
    } else {
      wait_async_le0();
    }
    __syncthreads();

    v16bf af = load_a_frag(&sA[buf][(wm * 16 + l16) * 32 + half * 16]);
    const int k = kt * 32;
#pragma unroll
    for (int j = 0; j < 4; ++j) {
      const __bf16* brow =
          Bm + (size_t)(nbase + j * 16 + l16) * Kdim + k + half * 16;
      v16bf bf = load_b_frag(brow);
      acc[j] = __builtin_amdgcn_wmma_f32_16x16x32_bf16(
          false, af, false, bf, (short)0, acc[j], false, false);
    }
    __syncthreads();                     // everyone done reading buf
  }
  // C/D layout: vgpr r -> M = r + half*8, lane -> N = l16
#pragma unroll
  for (int j = 0; j < 4; ++j)
#pragma unroll
    for (int r = 0; r < 8; ++r)
      C[(size_t)(mblock + wm * 16 + half * 8 + r) * Ndim +
        (nbase + j * 16 + l16)] = acc[j][r];
}

// ---------------------------------------------------------------------------
// RoPE + head-major permute: src (B*T, nheads*128) f32 -> dst (B,nheads,T,128) bf16
__global__ void rope_permute(const float* __restrict__ src,
                             __bf16* __restrict__ dst, int nheads) {
  int idx = blockIdx.x * blockDim.x + threadIdx.x;      // ((b*T+t)*nh + h)*128 + d
  int d   = idx & (HD - 1);
  int tmp = idx >> 7;
  int h   = tmp % nheads;
  int bt  = tmp / nheads;
  int t   = bt & (TT - 1);
  int b   = bt >> 11;
  float x  = src[idx];
  int   dp = d ^ 64;
  float xp = src[idx - d + dp];
  float sgn = (d < 64) ? -1.0f : 1.0f;
  // inv_freq = theta^(-2*(d%64)/128) = exp(-(d&63) * ln(10000)/64)
  float ang = (float)t * __expf(-(float)(d & 63) * 0.14391156f);
  float c = __cosf(ang), s = __sinf(ang);
  float val = x * c + sgn * xp * s;
  dst[(((size_t)b * nheads + h) * TT + t) * HD + d] = (__bf16)val;
}

// V transpose: src (B*T, NKV*128) f32 -> dst (B,NKV,128,T) bf16   (V^T per head)
__global__ void v_transpose(const float* __restrict__ src, __bf16* __restrict__ dst) {
  int idx = blockIdx.x * blockDim.x + threadIdx.x;      // over B*T*NKV*128
  int d   = idx & (HD - 1);
  int tmp = idx >> 7;
  int h   = tmp % NKV;
  int bt  = tmp / NKV;
  int t   = bt & (TT - 1);
  int b   = bt >> 11;
  dst[(((size_t)b * NKV + h) * HD + d) * TT + t] = (__bf16)src[idx];
}

// ---------------------------------------------------------------------------
// Causal GQA flash attention via WMMA, S^T formulation (all stats lane-local).
// Q (B,NH,T,128) bf16 ; K (B,NKV,T,128) bf16 ; Vt (B,NKV,128,T) bf16
// O (B*T, NH*128) bf16.  Grid: (B*NH, T/128), 256 thr = 8 waves x 16 q-rows.
__global__ __launch_bounds__(256)
void attn_fwd(const __bf16* __restrict__ Q, const __bf16* __restrict__ K,
              const __bf16* __restrict__ Vt, __bf16* __restrict__ O) {
  const int lane = threadIdx.x & 31;
  const int wave = threadIdx.x >> 5;
  const int half = lane >> 4;
  const int l16  = lane & 15;
  const int bh = blockIdx.x;
  const int b  = bh / NH, h = bh % NH;
  const int hk = h / (NH / NKV);
  const int qbase = blockIdx.y * 128 + wave * 16;
  const int qrow  = qbase + l16;                 // this lane's query index

  const __bf16* qptr = Q  + ((size_t)bh * TT + qbase) * HD;
  const __bf16* kptr = K  + ((size_t)(b * NKV + hk) * TT) * HD;
  const __bf16* vptr = Vt + ((size_t)(b * NKV + hk) * HD) * TT;

  // Q as B-operand fragments (N = q row, K = head dim), 4 slices of 32
  v16bf qf[4];
#pragma unroll
  for (int j = 0; j < 4; ++j)
    qf[j] = load_b_frag(qptr + (size_t)l16 * HD + j * 32 + half * 16);

  v8f   o[8] = {};                 // O^T accum: 8 d-tiles of 16
  float mrow = -1e30f, lrow = 0.0f;
  const float scale = 0.08838834764831845f;      // 1/sqrt(128)

  const int kend = qbase + 16;                   // causal bound (exclusive)
  for (int kb = 0; kb < kend; kb += 32) {
    // S^T tile (32 k-rows x 16 q-cols) = K_tile (A) x Q^T (B)
    v8f st[2] = {};
#pragma unroll
    for (int ka = 0; ka < 2; ++ka) {
      const __bf16* kr = kptr + (size_t)(kb + ka * 16 + l16) * HD + half * 16;
#pragma unroll
      for (int j = 0; j < 4; ++j) {
        v16bf kf = load_a_frag(kr + j * 32);
        st[ka] = __builtin_amdgcn_wmma_f32_16x16x32_bf16(
            false, kf, false, qf[j], (short)0, st[ka], false, false);
      }
    }
    // scale + causal mask; lane-local row max (k = kb + ka*16 + half*8 + r)
    float tmax = -1e30f;
#pragma unroll
    for (int ka = 0; ka < 2; ++ka)
#pragma unroll
      for (int r = 0; r < 8; ++r) {
        int   kidx = kb + ka * 16 + half * 8 + r;
        float s = st[ka][r] * scale;
        s = (kidx > qrow) ? -1e30f : s;
        st[ka][r] = s;
        tmax = fmaxf(tmax, s);
      }
    tmax = fmaxf(tmax, __shfl_xor(tmax, 16, 32));
    float mnew  = fmaxf(mrow, tmax);
    float alpha = __expf(mrow - mnew);
    mrow = mnew;

    // P = exp(S^T - m), row sums, pack P^T into a B-operand fragment.
    float ex0[8], ex1[8], rsum = 0.0f;
#pragma unroll
    for (int r = 0; r < 8; ++r) {
      ex0[r] = __expf(st[0][r] - mnew);
      ex1[r] = __expf(st[1][r] - mnew);
      rsum  += ex0[r] + ex1[r];
    }
    rsum += __shfl_xor(rsum, 16, 32);
    lrow  = lrow * alpha + rsum;

    // B-frag needs: lanes 0-15 -> k 0..15 ; lanes 16-31 -> k 16..31.
    // C layout holds k{0-7,16-23} on lanes<16 and k{8-15,24-31} on lanes>=16,
    // so swap the missing halves across lane^16 (same q on both halves).
    v16bf pb;
#pragma unroll
    for (int r = 0; r < 8; ++r) {
      float p0 = __shfl_xor(ex0[r], 16, 32);   // partner's St0 values
      float p1 = __shfl_xor(ex1[r], 16, 32);   // partner's St1 values
      pb[r]     = (__bf16)(half ? p1 : ex0[r]);   // k = r      / 16+r
      pb[8 + r] = (__bf16)(half ? ex1[r] : p0);   // k = 8+r    / 24+r
    }

    // O^T = O^T*alpha + V^T(A) x P^T(B)
#pragma unroll
    for (int i = 0; i < 8; ++i) {
#pragma unroll
      for (int r = 0; r < 8; ++r) o[i][r] *= alpha;
      v16bf vf = load_a_frag(vptr + (size_t)(i * 16 + l16) * TT + kb + half * 16);
      o[i] = __builtin_amdgcn_wmma_f32_16x16x32_bf16(
          false, vf, false, pb, (short)0, o[i], false, false);
    }
  }

  float inv = 1.0f / lrow;
  __bf16* optr = O + ((size_t)b * TT + qrow) * (NH * HD) + h * HD;
#pragma unroll
  for (int i = 0; i < 8; ++i)
#pragma unroll
    for (int r = 0; r < 8; ++r)
      optr[i * 16 + half * 8 + r] = (__bf16)(o[i][r] * inv);
}

// ---------------------------------------------------------------------------
extern "C" void kernel_launch(void* const* d_in, const int* in_sizes, int n_in,
                              void* d_out, int out_size, void* d_ws, size_t ws_size,
                              hipStream_t stream) {
  const float* x   = (const float*)d_in[0];
  const float* wq  = (const float*)d_in[1];
  const float* wkv = (const float*)d_in[2];
  const float* wk  = (const float*)d_in[3];
  const float* wv  = (const float*)d_in[4];
  const float* wo  = (const float*)d_in[5];
  float* out = (float*)d_out;

  char* w = (char*)d_ws;
  auto alloc = [&](size_t bytes) -> void* {
    void* p = (void*)w;
    w += (bytes + 255) & ~(size_t)255;
    return p;
  };
  __bf16* xb   = (__bf16*)alloc((size_t)MR * HID * 2);
  __bf16* wqb  = (__bf16*)alloc((size_t)HID * HID * 2);
  __bf16* wkvb = (__bf16*)alloc((size_t)LAT * HID * 2);
  __bf16* wkb  = (__bf16*)alloc((size_t)KVD * LAT * 2);
  __bf16* wvb  = (__bf16*)alloc((size_t)KVD * LAT * 2);
  __bf16* wob  = (__bf16*)alloc((size_t)HID * HID * 2);
  float*  qf   = (float*) alloc((size_t)MR * HID * 4);
  float*  kvf  = (float*) alloc((size_t)MR * LAT * 4);
  __bf16* kvb  = (__bf16*)alloc((size_t)MR * LAT * 2);
  float*  kf   = (float*) alloc((size_t)MR * KVD * 4);
  float*  vf   = (float*) alloc((size_t)MR * KVD * 4);
  __bf16* qb   = (__bf16*)alloc((size_t)BB * NH  * TT * HD * 2);
  __bf16* kb   = (__bf16*)alloc((size_t)BB * NKV * TT * HD * 2);
  __bf16* vtb  = (__bf16*)alloc((size_t)BB * NKV * HD * TT * 2);
  __bf16* aob  = (__bf16*)alloc((size_t)MR * HID * 2);

  auto cvt = [&](const float* s, __bf16* d, int n) {
    cvt_f32_to_bf16<<<(n + 255) / 256, 256, 0, stream>>>(s, d, n);
  };
  // 1) casts to bf16
  cvt(x,   xb,   MR * HID);
  cvt(wq,  wqb,  HID * HID);
  cvt(wkv, wkvb, LAT * HID);
  cvt(wk,  wkb,  KVD * LAT);
  cvt(wv,  wvb,  KVD * LAT);
  cvt(wo,  wob,  HID * HID);

  // 2) projections
  gemm_bf16_nt<<<dim3(MR / 32, HID / 256), 256, 0, stream>>>(xb, wqb,  qf,  MR, HID, HID);
  gemm_bf16_nt<<<dim3(MR / 32, LAT / 256), 256, 0, stream>>>(xb, wkvb, kvf, MR, LAT, HID);
  cvt(kvf, kvb, MR * LAT);
  gemm_bf16_nt<<<dim3(MR / 32, KVD / 256), 256, 0, stream>>>(kvb, wkb, kf, MR, KVD, LAT);
  gemm_bf16_nt<<<dim3(MR / 32, KVD / 256), 256, 0, stream>>>(kvb, wvb, vf, MR, KVD, LAT);

  // 3) RoPE + head-major permute + V transpose
  rope_permute<<<(MR * NH  * HD) / 256, 256, 0, stream>>>(qf, qb, NH);
  rope_permute<<<(MR * NKV * HD) / 256, 256, 0, stream>>>(kf, kb, NKV);
  v_transpose <<<(MR * NKV * HD) / 256, 256, 0, stream>>>(vf, vtb);

  // 4) flash attention
  attn_fwd<<<dim3(BB * NH, TT / 128), 256, 0, stream>>>(qb, kb, vtb, aob);

  // 5) output projection -> f32 d_out
  gemm_bf16_nt<<<dim3(MR / 32, HID / 256), 256, 0, stream>>>(aob, wob, out, MR, HID, HID);

  (void)in_sizes; (void)n_in; (void)out_size; (void)ws_size;
}
